// SparseEquivariantAttentionPairBias_43671227466529
// MI455X (gfx1250) — matrix operand order repaired and checked by reading, not compiled, not run on hardware
//
#include <hip/hip_runtime.h>
#include <hip/hip_bf16.h>

typedef __attribute__((ext_vector_type(16))) __bf16 v16bf;
typedef __attribute__((ext_vector_type(8)))  float  v8f;
typedef __attribute__((ext_vector_type(2)))  __bf16 bf2;
typedef __attribute__((ext_vector_type(4)))  unsigned int v4u;
typedef __attribute__((ext_vector_type(8)))  int v8i_t;
typedef __attribute__((ext_vector_type(4)))  int v4i_t;

#define NN  1024
#define DD  1024
#define NH  16      // heads
#define HD  64      // head dim
#define CZ  128
#define KK  512     // top-k

#if defined(__has_builtin)
#  if __has_builtin(__builtin_amdgcn_tensor_load_to_lds)
#    define HAVE_TDM 1
#  endif
#endif

// ---- WMMA fragment index helpers (wave32, 16-bit A/B, per CDNA5 ISA) ----
// A 16x32 bf16: lane<16 holds row=lane, K {0..7,16..23}; lane>=16 same row, K {8..15,24..31}.
__device__ __forceinline__ int a_kbase(int i, int grp) { return ((i & 4) << 2) + grp * 8 + ((i & 3) << 1); }
// B 32x16 bf16: col = lane&15; lanes 0-15 K {0..7,16..23}, lanes 16-31 K {8..15,24..31}, 2 per vgpr.
__device__ __forceinline__ int b_kbase(int i, int grp) { return ((i & 4) << 2) + grp * 8 + ((i & 3) << 1); }
// C/D f32 16x16: col = lane&15, row = vgpr + 8*(lane>>4).

#define WMMA_BF16(a, b, c) \
  __builtin_amdgcn_wmma_f32_16x16x32_bf16(false, (a), false, (b), (short)0, (c), false, false)

// =====================================================================
// K1: q = s*Wq^T + bq ; k = k_in*Wk^T ; v = k_in*Wv^T ; g = sigmoid(s*Wg^T)
// One wave -> 16x64 output tile: A fragment reused across 4 B tiles.
// =====================================================================
__global__ __launch_bounds__(128) void k_proj(
    const float* __restrict__ s, const float* __restrict__ k_in,
    const float* __restrict__ Wq, const float* __restrict__ bq,
    const float* __restrict__ Wk, const float* __restrict__ Wv,
    const float* __restrict__ Wg,
    __bf16* __restrict__ qb, __bf16* __restrict__ kb,
    __bf16* __restrict__ vb, float* __restrict__ gf)
{
  const int lane = threadIdx.x, grp = lane >> 4, lo = lane & 15;
  const int nt0 = blockIdx.x * 4;                  // first output-col tile (e)
  const int mt  = blockIdx.y * 4 + threadIdx.y;    // output-row tile (n)
  const int id  = blockIdx.z;                      // 0=q 1=k 2=v 3=g
  const float* A = (id == 1 || id == 2) ? k_in : s;
  const float* W = (id == 0) ? Wq : (id == 1) ? Wk : (id == 2) ? Wv : Wg;
  const int arow = mt * 16 + lo;

  v8f c[4] = {};
  for (int cc = 0; cc < DD; cc += 32) {
    v16bf a;
#pragma unroll
    for (int i = 0; i < 8; i++) {
      const float2 av = *(const float2*)(A + (size_t)arow * DD + cc + a_kbase(i, grp));
      a[2 * i] = (__bf16)av.x; a[2 * i + 1] = (__bf16)av.y;
    }
#pragma unroll
    for (int t = 0; t < 4; t++) {
      const int wrow = (nt0 + t) * 16 + lo;
      v16bf b;
#pragma unroll
      for (int i = 0; i < 8; i++) {
        const float2 wv = *(const float2*)(W + (size_t)wrow * DD + cc + b_kbase(i, grp));
        b[2 * i] = (__bf16)wv.x; b[2 * i + 1] = (__bf16)wv.y;
      }
      c[t] = WMMA_BF16(a, b, c[t]);
    }
  }

#pragma unroll
  for (int t = 0; t < 4; t++) {
    const int col = (nt0 + t) * 16 + lo;
    const float bqv = (id == 0) ? bq[col] : 0.0f;
#pragma unroll
    for (int iv = 0; iv < 8; iv++) {
      const size_t row = (size_t)(mt * 16 + grp * 8 + iv);
      const float vv = c[t][iv];
      if      (id == 0) qb[row * DD + col] = (__bf16)(vv + bqv);
      else if (id == 1) kb[row * DD + col] = (__bf16)vv;
      else if (id == 2) vb[row * DD + col] = (__bf16)vv;
      else              gf[row * DD + col] = __builtin_amdgcn_rcpf(1.0f + __expf(-vv));
    }
  }
}

// =====================================================================
// K2: scores[h,i,j] = (q.k)/8 + mask * SphHarm(pos_i - pos_j) . Wh[:,h]
// =====================================================================
__global__ __launch_bounds__(32) void k_scores(
    const __bf16* __restrict__ qb, const __bf16* __restrict__ kb,
    const float* __restrict__ pos, const unsigned char* __restrict__ mask,
    const float* __restrict__ Wh, float* __restrict__ scores)
{
  const int lane = threadIdx.x, grp = lane >> 4, lo = lane & 15;
  const int jt = blockIdx.x, it = blockIdx.y, h = blockIdx.z;
  const int qrow = it * 16 + lo;
  const int krow = jt * 16 + lo;
  const int cbase = h * HD;

  v8f c = {};
#pragma unroll
  for (int kkk = 0; kkk < 2; kkk++) {
    const int cc = cbase + kkk * 32;
    v16bf a, b;
#pragma unroll
    for (int i = 0; i < 8; i++) {
      const bf2 ap = *(const bf2*)(qb + (size_t)qrow * DD + cc + a_kbase(i, grp));
      a[2 * i] = ap[0]; a[2 * i + 1] = ap[1];
      const bf2 bp = *(const bf2*)(kb + (size_t)krow * DD + cc + b_kbase(i, grp));
      b[2 * i] = bp[0]; b[2 * i + 1] = bp[1];
    }
    c = WMMA_BF16(a, b, c);
  }

  const float c0  = 0.28209479177387814f;
  const float c1  = 0.48860251190291992f;
  const float c2a = 1.09254843059207907f;
  const float c2b = 0.77254840404637897f;
  const float c2c = 0.31539156525252005f;
  float wh[9];
#pragma unroll
  for (int k9 = 0; k9 < 9; k9++) wh[k9] = Wh[k9 * NH + h];

  const int j = jt * 16 + lo;
  const float pjx = pos[j * 3 + 0], pjy = pos[j * 3 + 1], pjz = pos[j * 3 + 2];
#pragma unroll
  for (int iv = 0; iv < 8; iv++) {
    const int i = it * 16 + grp * 8 + iv;
    const float rx = pos[i * 3 + 0] - pjx;
    const float ry = pos[i * 3 + 1] - pjy;
    const float rz = pos[i * 3 + 2] - pjz;
    const float r = fmaxf(sqrtf(rx * rx + ry * ry + rz * rz), 1e-8f);
    const float rinv = __builtin_amdgcn_rcpf(r);
    const float x = rx * rinv, y = ry * rinv, z = rz * rinv;
    float hb = c0 * wh[0] + c1 * (y * wh[1] + z * wh[2] + x * wh[3]) +
               c2a * (x * y * wh[4] + y * z * wh[5] + x * z * wh[7]) +
               c2c * (3.0f * z * z - 1.0f) * wh[6] + c2b * (x * x - y * y) * wh[8];
    hb *= mask[(size_t)i * NN + j] ? 1.0f : 0.0f;
    scores[((size_t)h * NN + i) * NN + j] = c[iv] * 0.125f + hb;
  }
}

// =====================================================================
// K3: scores[h,i,j] += LayerNorm(z[i,j,:]) @ Wz[:,h]   (streams 512MB of z)
// TDM streams one 16x128 fp32 tile per wave into LDS (4 dword pad per row
// to avoid bank conflicts), then per-pair LN in registers + 4 WMMA steps.
// =====================================================================
__global__ __launch_bounds__(128) void k_zbias(
    const float* __restrict__ z, const float* __restrict__ ln_g,
    const float* __restrict__ ln_b, const float* __restrict__ Wz,
    float* __restrict__ scores)
{
  __shared__ float sg[CZ], sb[CZ];
#if defined(HAVE_TDM)
  __shared__ __align__(16) float ztile[4][16][CZ + 4];   // +4 dwords pad / row
#endif
  const int lane = threadIdx.x, grp = lane >> 4, lo = lane & 15;
  const int wid = threadIdx.y;
  const int jt = blockIdx.x * 4 + wid;
  const int i  = blockIdx.y;

#if defined(HAVE_TDM)
  {
    // Tensor DMA descriptor (CDNA5 ISA): 2D tile,
    // tile_dim0=128 (channels, 4B), tile_dim1=16 (pairs), row stride = CZ,
    // LDS pad: 4 dwords after every 128 dwords.
    const unsigned long long ga =
        (unsigned long long)(const void*)(z + ((size_t)i * NN + jt * 16) * CZ);
    const unsigned lds = (unsigned)(unsigned long long)(const void*)&ztile[wid][0][0];
    v4u g0;
    g0[0] = 1u;                                          // count=1 (valid user D#)
    g0[1] = lds;                                         // lds_addr
    g0[2] = (unsigned)(ga & 0xFFFFFFFFu);                // global_addr[31:0]
    g0[3] = (unsigned)((ga >> 32) & 0x01FFFFFFu) | (2u << 30);  // addr[56:32] | type=2
    v8i_t g1;
    g1[0] = (int)(0x00020000u                            // data_size=4B
                | (1u << 20)                             // pad_enable
                | (6u << 22)                             // pad_interval: 128 dwords
                | (3u << 25));                           // pad_amount: 4 dwords
    g1[1] = (int)((CZ & 0xFFFFu) << 16);                 // tensor_dim0[15:0]
    g1[2] = (int)(((unsigned)(NN * NN) & 0xFFFFu) << 16);// dim0[31:16]=0 | dim1[15:0]
    g1[3] = (int)((((unsigned)(NN * NN)) >> 16) | ((unsigned)CZ << 16)); // dim1[31:16] | tile_dim0
    g1[4] = 16;                                          // tile_dim1=16, tile_dim2=0
    g1[5] = CZ;                                          // tensor_dim0_stride[31:0]
    g1[6] = (int)((unsigned)CZ << 16);                   // stride0[47:32]=0 | stride1[15:0]
    g1[7] = 0;                                           // stride1[47:16]
    v4i_t g2 = {}, g3 = {};
#if __clang_major__ >= 23
    v8i_t g1b = {};
    __builtin_amdgcn_tensor_load_to_lds(g0, g1, g2, g3, g1b, 0);
#else
    __builtin_amdgcn_tensor_load_to_lds(g0, g1, g2, g3, 0);
#endif
  }
#endif

  const int t = wid * 32 + lane;
  if (t < CZ) { sg[t] = ln_g[t]; sb[t] = ln_b[t]; }
  __syncthreads();

#if defined(HAVE_TDM)
  __builtin_amdgcn_s_wait_tensorcnt(0);
  const float* zrow = &ztile[wid][lo][0];
#else
  const float* zrow = z + ((size_t)i * NN + jt * 16 + lo) * CZ;
  __builtin_prefetch(zrow, 0, 1);
#endif

  float zr[64];
  float sum = 0.0f, sq = 0.0f;
#pragma unroll
  for (int g8 = 0; g8 < 8; g8++) {
    const int cstart = (g8 >> 1) * 32 + (g8 & 1) * 16 + grp * 8;
    const float4 v0 = *(const float4*)(zrow + cstart);
    const float4 v1 = *(const float4*)(zrow + cstart + 4);
    zr[g8 * 8 + 0] = v0.x; zr[g8 * 8 + 1] = v0.y; zr[g8 * 8 + 2] = v0.z; zr[g8 * 8 + 3] = v0.w;
    zr[g8 * 8 + 4] = v1.x; zr[g8 * 8 + 5] = v1.y; zr[g8 * 8 + 6] = v1.z; zr[g8 * 8 + 7] = v1.w;
#pragma unroll
    for (int q = 0; q < 8; q++) { const float zz = zr[g8 * 8 + q]; sum += zz; sq += zz * zz; }
  }
  sum += __shfl_xor(sum, 16);
  sq  += __shfl_xor(sq, 16);
  const float mu  = sum * (1.0f / CZ);
  const float var = sq * (1.0f / CZ) - mu * mu;
  const float inv = rsqrtf(var + 1e-5f);

  v8f c = {};
#pragma unroll
  for (int kkk = 0; kkk < 4; kkk++) {
    v16bf a, b;
#pragma unroll
    for (int ii = 0; ii < 8; ii++) {
      const int cA = kkk * 32 + a_kbase(ii, grp);
      const int li = (kkk * 2 + (ii >> 2)) * 8 + ((ii & 3) << 1);
      a[2 * ii]     = (__bf16)((zr[li]     - mu) * inv * sg[cA]     + sb[cA]);
      a[2 * ii + 1] = (__bf16)((zr[li + 1] - mu) * inv * sg[cA + 1] + sb[cA + 1]);
      const int cB = kkk * 32 + b_kbase(ii, grp);
      b[2 * ii]     = (__bf16)Wz[(size_t)cB * NH + lo];
      b[2 * ii + 1] = (__bf16)Wz[(size_t)(cB + 1) * NH + lo];
    }
    c = WMMA_BF16(a, b, c);
  }

  // C: row m = pair (j), col n = head. Each (h,i,j) touched exactly once.
#pragma unroll
  for (int iv = 0; iv < 8; iv++) {
    const int jj = jt * 16 + grp * 8 + iv;
    const size_t idx = ((size_t)lo * NN + i) * NN + jj;
    scores[idx] += c[iv];
  }
}

// =====================================================================
// K4: per (h,i) row: exact top-512 (radix select + index-ordered ties) + softmax
// =====================================================================
__global__ __launch_bounds__(256) void k_topk_softmax(
    const float* __restrict__ scores, const unsigned char* __restrict__ mask,
    __bf16* __restrict__ attn)
{
  __shared__ unsigned keys[NN];
  __shared__ unsigned hist[256];
  __shared__ unsigned scn[256];
  __shared__ float    redf[256];
  __shared__ unsigned sel[2];

  const int i = blockIdx.x, h = blockIdx.y, t = threadIdx.x;
  const float* row = scores + ((size_t)h * NN + i) * NN;
  const unsigned char* mrow = mask + (size_t)i * NN;

  const float4 v4 = *(const float4*)(row + 4 * t);
  float lv[4] = {v4.x, v4.y, v4.z, v4.w};
#pragma unroll
  for (int p = 0; p < 4; p++) {
    if (!mrow[4 * t + p]) lv[p] = -__builtin_inff();
    const unsigned bbits = __float_as_uint(lv[p]);
    const unsigned u = bbits ^ ((unsigned)((int)bbits >> 31) | 0x80000000u);
    keys[4 * t + p] = ~u;   // ascending key order == descending float order
  }
  __syncthreads();

  unsigned want = KK, prefix = 0;
  for (int shift = 24; shift >= 0; shift -= 8) {
    hist[t] = 0; __syncthreads();
    const unsigned hm = (shift == 24) ? 0u : (0xFFFFFFFFu << (shift + 8));
#pragma unroll
    for (int p = 0; p < 4; p++) {
      const unsigned kv = keys[4 * t + p];
      if ((kv & hm) == prefix) atomicAdd(&hist[(kv >> shift) & 255], 1u);
    }
    __syncthreads();
    if (t == 0) {
      unsigned cum = 0;
      for (int bn = 0; bn < 256; bn++) {
        const unsigned hc = hist[bn];
        if (cum + hc >= want) { sel[0] = (unsigned)bn; sel[1] = want - cum; break; }
        cum += hc;
      }
    }
    __syncthreads();
    prefix |= sel[0] << shift; want = sel[1];
    __syncthreads();
  }
  const unsigned T = prefix;   // key of the 512th-ranked element

  unsigned lc = 0, tc = 0;
#pragma unroll
  for (int p = 0; p < 4; p++) {
    const unsigned kv = keys[4 * t + p];
    lc += (kv < T); tc += (kv == T);
  }
  scn[t] = lc; __syncthreads();
  for (int off = 128; off > 0; off >>= 1) { if (t < off) scn[t] += scn[t + off]; __syncthreads(); }
  const unsigned nLess = scn[0]; __syncthreads();
  const unsigned R = KK - nLess;   // how many tied elements to take (index order)

  scn[t] = tc; __syncthreads();
  for (int off = 1; off < 256; off <<= 1) {
    const unsigned vp = (t >= off) ? scn[t - off] : 0u;
    __syncthreads();
    scn[t] += vp;
    __syncthreads();
  }
  const unsigned excl = scn[t] - tc;

  bool selp[4];
  float mx = -__builtin_inff();
  unsigned tr = excl;
#pragma unroll
  for (int p = 0; p < 4; p++) {
    const unsigned kv = keys[4 * t + p];
    bool sflag = (kv < T);
    if (kv == T) { sflag = (tr < R); tr++; }
    selp[p] = sflag;
    if (sflag) mx = fmaxf(mx, lv[p]);
  }
  redf[t] = mx; __syncthreads();
  for (int off = 128; off > 0; off >>= 1) { if (t < off) redf[t] = fmaxf(redf[t], redf[t + off]); __syncthreads(); }
  mx = redf[0]; __syncthreads();

  float ex[4]; float se = 0.0f;
#pragma unroll
  for (int p = 0; p < 4; p++) {
    ex[p] = selp[p] ? __expf(lv[p] - mx) : 0.0f;
    se += ex[p];
  }
  redf[t] = se; __syncthreads();
  for (int off = 128; off > 0; off >>= 1) { if (t < off) redf[t] += redf[t + off]; __syncthreads(); }
  const float invs = __builtin_amdgcn_rcpf(redf[0]);

  __bf16* arow = attn + ((size_t)h * NN + i) * NN + 4 * t;
#pragma unroll
  for (int p = 0; p < 4; p++) arow[p] = (__bf16)(ex[p] * invs);
}

// =====================================================================
// K5: o[i, h*64+d] = sum_j attn[h,i,j] * v[j, h*64+d]
// One wave -> 16 rows x full head dim (4 tiles), A fragment reused 4x.
// =====================================================================
__global__ __launch_bounds__(32) void k_attnv(
    const __bf16* __restrict__ attn, const __bf16* __restrict__ vb,
    float* __restrict__ of)
{
  const int lane = threadIdx.x, grp = lane >> 4, lo = lane & 15;
  const int it = blockIdx.x, h = blockIdx.y;
  const __bf16* ap = attn + ((size_t)h * NN + it * 16 + lo) * NN;

  v8f c[4] = {};
  for (int jc = 0; jc < NN; jc += 32) {
    v16bf a;
#pragma unroll
    for (int ii = 0; ii < 8; ii++) {
      const bf2 av = *(const bf2*)(ap + jc + a_kbase(ii, grp));
      a[2 * ii] = av[0]; a[2 * ii + 1] = av[1];
    }
#pragma unroll
    for (int t = 0; t < 4; t++) {
      const int dcol = h * HD + t * 16 + lo;
      v16bf b;
#pragma unroll
      for (int ii = 0; ii < 8; ii++) {
        const int kb2 = jc + b_kbase(ii, grp);
        b[2 * ii]     = vb[(size_t)kb2 * DD + dcol];
        b[2 * ii + 1] = vb[(size_t)(kb2 + 1) * DD + dcol];
      }
      c[t] = WMMA_BF16(a, b, c[t]);
    }
  }
#pragma unroll
  for (int t = 0; t < 4; t++)
#pragma unroll
    for (int iv = 0; iv < 8; iv++)
      of[(size_t)(it * 16 + grp * 8 + iv) * DD + h * HD + t * 16 + lo] = c[t][iv];
}

// =====================================================================
// K6: out[n,e] = sum_d (g[n,d]*o[n,d]) * Wo[e,d]  (16x64 tile per wave)
// =====================================================================
__global__ __launch_bounds__(128) void k_out(
    const float* __restrict__ gf, const float* __restrict__ of,
    const float* __restrict__ Wo, float* __restrict__ out)
{
  const int lane = threadIdx.x, grp = lane >> 4, lo = lane & 15;
  const int nt0 = blockIdx.x * 4;
  const int mt  = blockIdx.y * 4 + threadIdx.y;
  const int arow = mt * 16 + lo;

  v8f c[4] = {};
  for (int cc = 0; cc < DD; cc += 32) {
    v16bf a;
#pragma unroll
    for (int ii = 0; ii < 8; ii++) {
      const int ka = cc + a_kbase(ii, grp);
      const float2 gv = *(const float2*)(gf + (size_t)arow * DD + ka);
      const float2 ov = *(const float2*)(of + (size_t)arow * DD + ka);
      a[2 * ii] = (__bf16)(gv.x * ov.x); a[2 * ii + 1] = (__bf16)(gv.y * ov.y);
    }
#pragma unroll
    for (int t = 0; t < 4; t++) {
      const int wrow = (nt0 + t) * 16 + lo;
      v16bf b;
#pragma unroll
      for (int ii = 0; ii < 8; ii++) {
        const float2 wv = *(const float2*)(Wo + (size_t)wrow * DD + cc + b_kbase(ii, grp));
        b[2 * ii] = (__bf16)wv.x; b[2 * ii + 1] = (__bf16)wv.y;
      }
      c[t] = WMMA_BF16(a, b, c[t]);
    }
  }
#pragma unroll
  for (int t = 0; t < 4; t++)
#pragma unroll
    for (int iv = 0; iv < 8; iv++)
      out[(size_t)(mt * 16 + grp * 8 + iv) * DD + (nt0 + t) * 16 + lo] = c[t][iv];
}

// =====================================================================
extern "C" void kernel_launch(void* const* d_in, const int* in_sizes, int n_in,
                              void* d_out, int out_size, void* d_ws, size_t ws_size,
                              hipStream_t stream)
{
  const float* s     = (const float*)d_in[0];
  const float* z     = (const float*)d_in[1];
  const unsigned char* mask = (const unsigned char*)d_in[2];
  const float* k_in  = (const float*)d_in[3];
  const float* pos   = (const float*)d_in[4];
  const float* Wq    = (const float*)d_in[5];
  const float* bq    = (const float*)d_in[6];
  const float* Wk    = (const float*)d_in[7];
  const float* Wv    = (const float*)d_in[8];
  const float* Wg    = (const float*)d_in[9];
  const float* Wh    = (const float*)d_in[10];
  const float* ln_g  = (const float*)d_in[11];
  const float* ln_b  = (const float*)d_in[12];
  const float* Wz    = (const float*)d_in[13];
  const float* Wo    = (const float*)d_in[14];
  float* out = (float*)d_out;

  char* ws = (char*)d_ws;
  __bf16* qb     = (__bf16*)(ws + (size_t)0);
  __bf16* kb     = (__bf16*)(ws + ((size_t)2 << 20));
  __bf16* vb     = (__bf16*)(ws + ((size_t)4 << 20));
  float*  gf     = (float*)(ws + ((size_t)6 << 20));
  float*  of     = (float*)(ws + ((size_t)10 << 20));
  float*  scores = (float*)(ws + ((size_t)14 << 20));
  __bf16* attn   = (__bf16*)(ws + ((size_t)78 << 20));

  const dim3 b128(32, 4, 1);

  k_proj<<<dim3((DD / 16) / 4, (NN / 16) / 4, 4), b128, 0, stream>>>(
      s, k_in, Wq, bq, Wk, Wv, Wg, qb, kb, vb, gf);

  k_scores<<<dim3(NN / 16, NN / 16, NH), dim3(32, 1, 1), 0, stream>>>(
      qb, kb, pos, mask, Wh, scores);

  k_zbias<<<dim3((NN / 16) / 4, NN, 1), b128, 0, stream>>>(
      z, ln_g, ln_b, Wz, scores);

  k_topk_softmax<<<dim3(NN, NH, 1), dim3(256, 1, 1), 0, stream>>>(
      scores, mask, attn);

  k_attnv<<<dim3(NN / 16, NH, 1), dim3(32, 1, 1), 0, stream>>>(
      attn, vb, of);

  k_out<<<dim3((DD / 16) / 4, (NN / 16) / 4, 1), b128, 0, stream>>>(
      gf, of, Wo, out);
}
